// NMRShiftModel_30279519437525
// MI455X (gfx1250) — matrix operand order repaired and checked by reading, not compiled, not run on hardware
//
#include <hip/hip_runtime.h>
#include <hip/hip_bf16.h>

typedef __attribute__((ext_vector_type(16))) __bf16 v16bf;
typedef __attribute__((ext_vector_type(8)))  __bf16 v8bf;
typedef __attribute__((ext_vector_type(8)))  float  v8f;

// explicit global-address-space pointer types (force global_load_* codegen)
typedef const __bf16 __attribute__((address_space(1)))* gcbf;
typedef const v8bf   __attribute__((address_space(1)))* gcv8;
typedef float        __attribute__((address_space(1)))* gf32;

#define N_NODES  25000
#define NPAD     25024      // 1564 * 16 ; 1564 = 391 * 4 M-tiles
#define N_EDGES  400000
#define N_GRAPHS 1000
#define HID      185
#define IN_CH    9
#define KP       192        // padded hidden stride (12 tiles of 16)
#define NTILES   12         // KP / 16
#define JKW      768        // padded JK concat width (4*185 = 740 -> 768)
#define MT       4          // M-tiles per wave in the GEMM
#define BN_EPS   1e-5f

static inline int cdiv(int a, int b) { return (a + b - 1) / b; }

// ---------------------------------------------------------------------------
// degree / normalization
// ---------------------------------------------------------------------------
__global__ void k_init_deg(float* deg) {
    int n = blockIdx.x * blockDim.x + threadIdx.x;
    if (n < NPAD) deg[n] = (n < N_NODES) ? 1.0f : 0.0f;   // self-loop counts 1
}

__global__ void k_accum_deg(const int* dst, float* deg) {
    int e = blockIdx.x * blockDim.x + threadIdx.x;
    if (e < N_EDGES) atomicAdd(&deg[dst[e]], 1.0f);
}

__global__ void k_compute_dis(const float* deg, float* dis) {
    int n = blockIdx.x * blockDim.x + threadIdx.x;
    if (n < NPAD) {
        float d = deg[n];
        dis[n] = (n < N_NODES && d > 0.0f) ? rsqrtf(d) : 0.0f;
    }
}

// ---------------------------------------------------------------------------
// staging conversions
// ---------------------------------------------------------------------------
// grid = NPAD blocks, KP threads
__global__ void k_convert_x(const float* __restrict__ x, __bf16* __restrict__ Abf) {
    int n = blockIdx.x, c = threadIdx.x;
    float v = (n < N_NODES && c < IN_CH) ? x[n * IN_CH + c] : 0.0f;
    Abf[(size_t)n * KP + c] = (__bf16)v;
}

// W row-major [rows x cols] (f32) -> packed bf16 B-fragments.
// pack[((t*kchunks + kc)*32 + lane)*16 + j] = W[kc*32 + (lane>>4)*16 + j, t*16 + (lane&15)]
__global__ void k_pack_w(const float* __restrict__ W, __bf16* __restrict__ pack,
                         int rows, int cols, int kchunks) {
    int idx = blockIdx.x * blockDim.x + threadIdx.x;
    if (idx >= NTILES * kchunks * 32 * 16) return;
    int j    = idx & 15;
    int lane = (idx >> 4) & 31;
    int rest = idx >> 9;
    int kc   = rest % kchunks;
    int t    = rest / kchunks;
    int k = kc * 32 + (lane >> 4) * 16 + j;
    int n = t * 16 + (lane & 15);
    float v = (k < rows && n < cols) ? W[k * cols + n] : 0.0f;
    pack[idx] = (__bf16)v;
}

// ---------------------------------------------------------------------------
// WMMA GEMM: C[M x KP](f32) = A[M x LDA](bf16) @ Bpack (fragment-packed bf16)
// one wave per (4 M-tiles x 1 N-tile); grid = (M/64, KP/16)
// KCH and LDA are compile-time: all strides become immediate offsets, the K
// loop is fully unrolled and the scheduler hoists loads across WMMAs.
// ---------------------------------------------------------------------------
__device__ inline v16bf mk16(v8bf lo, v8bf hi) {
    return __builtin_shufflevector(lo, hi,
               0, 1, 2, 3, 4, 5, 6, 7, 8, 9, 10, 11, 12, 13, 14, 15);
}

template <int KCH, int LDA>
__global__ __launch_bounds__(32) void k_gemm_wmma(const __bf16* __restrict__ A,
                                                  const __bf16* __restrict__ Bpack,
                                                  float* __restrict__ C) {
    const int lane = threadIdx.x;
    const int tn   = blockIdx.y;
    const int tm0  = blockIdx.x * MT;

    const int mA  = lane & 15;
    const int kbA = (lane >> 4) * 16;       // A: K sub-offset per lane half

    gcbf abase = (gcbf)A + (size_t)(tm0 * 16 + mA) * LDA + kbA;
    gcbf bbase = (gcbf)Bpack + ((size_t)(tn * KCH) * 32 + lane) * 16;

    v8f zero = {};
    v8f acc[MT];
#pragma unroll
    for (int m = 0; m < MT; ++m) acc[m] = zero;

#pragma unroll
    for (int kc = 0; kc < KCH; ++kc) {
        v8bf b0 = *(gcv8)(bbase + kc * 512);
        v8bf b1 = *(gcv8)(bbase + kc * 512 + 8);
        v16bf bfrag = mk16(b0, b1);
#pragma unroll
        for (int m = 0; m < MT; ++m) {
            v8bf a0 = *(gcv8)(abase + m * 16 * LDA + kc * 32);
            v8bf a1 = *(gcv8)(abase + m * 16 * LDA + kc * 32 + 8);
            acc[m] = __builtin_amdgcn_wmma_f32_16x16x32_bf16(
                         false, mk16(a0, a1), false, bfrag, (short)0, acc[m], false, false);
        }
    }

    // C/D layout: lane = col (mod 16); VGPR r -> row (lane>=16 ? 8 : 0) + r
    const int rsub = (lane >> 4) * 8;
    const int col  = tn * 16 + (lane & 15);
    gf32 cbase = (gf32)C + (size_t)(tm0 * 16 + rsub) * KP + col;
#pragma unroll
    for (int m = 0; m < MT; ++m)
#pragma unroll
        for (int r = 0; r < 8; ++r)
            cbase[m * 16 * KP + r * KP] = acc[m][r];
}

// ---------------------------------------------------------------------------
// aggregation: Hagg = self-loop term, then edge scatter with f32 atomics
// ---------------------------------------------------------------------------
// grid = NPAD blocks, KP threads
__global__ void k_self_init(const float* __restrict__ Hpre, const float* __restrict__ dis,
                            float* __restrict__ Hagg) {
    int n = blockIdx.x, c = threadIdx.x;
    float d = dis[n];
    size_t idx = (size_t)n * KP + c;
    Hagg[idx] = Hpre[idx] * d * d;
}

__global__ void k_edge_scatter(const int* __restrict__ src, const int* __restrict__ dst,
                               const float* __restrict__ dis,
                               const float* __restrict__ Hpre, float* __restrict__ Hagg) {
    int e = (blockIdx.x * blockDim.x + threadIdx.x) >> 5;
    int lane = threadIdx.x & 31;
    if (e >= N_EDGES) return;
    int s = src[e], d = dst[e];
    const float* hp = Hpre + (size_t)s * KP;
    float* ha = Hagg + (size_t)d * KP;
    __builtin_prefetch(hp + lane, 0, 1);     // global_prefetch_b8: pull src row toward L0/L2
    float coef = dis[s] * dis[d];
#pragma unroll
    for (int j = 0; j < 6; ++j) {
        int c = lane + 32 * j;               // covers 0..191 (pads carry 0)
        atomicAdd(ha + c, hp[c] * coef);
    }
}

// BN (eval) + ReLU, emit next-layer bf16 input + JK concat slice
// grid = NPAD blocks, KP threads
__global__ void k_bn_relu_emit(const float* __restrict__ Hagg, const float* __restrict__ bias,
                               const float* __restrict__ gamma, const float* __restrict__ beta,
                               const float* __restrict__ mean, const float* __restrict__ var,
                               __bf16* __restrict__ Abf, __bf16* __restrict__ Xcat, int xoff) {
    int n = blockIdx.x, c = threadIdx.x;
    size_t idx = (size_t)n * KP + c;
    if (c < HID) {
        float v = Hagg[idx] + bias[c];
        v = (v - mean[c]) * rsqrtf(var[c] + BN_EPS) * gamma[c] + beta[c];
        v = fmaxf(v, 0.0f);
        __bf16 b = (__bf16)v;
        Abf[idx] = b;
        Xcat[(size_t)n * JKW + xoff + c] = b;
    } else {
        Abf[idx] = (__bf16)0.0f;
    }
}

__global__ void k_zero_xcat_pad(__bf16* Xcat) {
    int idx = blockIdx.x * blockDim.x + threadIdx.x;
    const int PADW = JKW - 4 * HID;          // 28
    if (idx >= NPAD * PADW) return;
    int n = idx / PADW, j = idx % PADW;
    Xcat[(size_t)n * JKW + 4 * HID + j] = (__bf16)0.0f;
}

// ---------------------------------------------------------------------------
// global_max_pool via order-preserving uint encoding + atomicMax
// ---------------------------------------------------------------------------
__global__ void k_init_pool(unsigned* keys) {
    int idx = blockIdx.x * blockDim.x + threadIdx.x;
    if (idx < N_GRAPHS * HID) keys[idx] = 0u;   // minimum key
}

__device__ inline unsigned f32_key(float f) {
    unsigned u = __float_as_uint(f);
    return (u >> 31) ? ~u : (u | 0x80000000u);
}

__global__ void k_pool_max(const int* __restrict__ batch, const float* __restrict__ Hjk,
                           const float* __restrict__ bjk, unsigned* __restrict__ keys) {
    int n = (blockIdx.x * blockDim.x + threadIdx.x) >> 5;
    int lane = threadIdx.x & 31;
    if (n >= N_NODES) return;
    int g = batch[n];
    const float* hrow = Hjk + (size_t)n * KP;
#pragma unroll
    for (int j = 0; j < 6; ++j) {
        int c = lane + 32 * j;
        if (c < HID) atomicMax(&keys[g * HID + c], f32_key(hrow[c] + bjk[c]));
    }
}

__global__ void k_final(const unsigned* __restrict__ keys, const float* __restrict__ Wout,
                        const float* __restrict__ bout, float* __restrict__ out) {
    int g = blockIdx.x * blockDim.x + threadIdx.x;
    if (g >= N_GRAPHS) return;
    float acc = bout[0];
    for (int c = 0; c < HID; ++c) {
        unsigned k = keys[g * HID + c];
        unsigned u = (k & 0x80000000u) ? (k ^ 0x80000000u) : ~k;
        acc += __uint_as_float(u) * Wout[c];
    }
    out[g] = acc;
}

// ---------------------------------------------------------------------------
extern "C" void kernel_launch(void* const* d_in, const int* in_sizes, int n_in,
                              void* d_out, int out_size, void* d_ws, size_t ws_size,
                              hipStream_t stream) {
    const float* x      = (const float*)d_in[0];
    const int*   ei     = (const int*)d_in[1];
    const int*   batch  = (const int*)d_in[2];
    const float* W0     = (const float*)d_in[3];
    const float* b0     = (const float*)d_in[4];
    const float* W_h    = (const float*)d_in[5];
    const float* b_h    = (const float*)d_in[6];
    const float* gam    = (const float*)d_in[7];
    const float* bet    = (const float*)d_in[8];
    const float* mean   = (const float*)d_in[9];
    const float* var    = (const float*)d_in[10];
    const float* W_jk   = (const float*)d_in[11];
    const float* b_jk   = (const float*)d_in[12];
    const float* W_out  = (const float*)d_in[13];
    const float* b_out  = (const float*)d_in[14];
    float* out = (float*)d_out;

    const int* src = ei;
    const int* dst = ei + N_EDGES;

    // workspace carve-out (all 256B-aligned)
    char* w = (char*)d_ws;
    auto carve = [&](size_t bytes) { char* p = w; w += (bytes + 255) & ~(size_t)255; return p; };
    float*    deg     = (float*)carve((size_t)NPAD * 4);
    float*    dis     = (float*)carve((size_t)NPAD * 4);
    __bf16*   Abf     = (__bf16*)carve((size_t)NPAD * KP * 2);
    __bf16*   Xcat    = (__bf16*)carve((size_t)NPAD * JKW * 2);
    float*    Hpre    = (float*)carve((size_t)NPAD * KP * 4);
    float*    Hagg    = (float*)carve((size_t)NPAD * KP * 4);
    __bf16*   W0pk    = (__bf16*)carve((size_t)NTILES * 1  * 512 * 2);
    __bf16*   Whpk    = (__bf16*)carve((size_t)3 * NTILES * 6 * 512 * 2);
    __bf16*   Wjkpk   = (__bf16*)carve((size_t)NTILES * 24 * 512 * 2);
    unsigned* keys    = (unsigned*)carve((size_t)N_GRAPHS * HID * 4);

    const int T = 256;
    // normalization
    k_init_deg<<<cdiv(NPAD, T), T, 0, stream>>>(deg);
    k_accum_deg<<<cdiv(N_EDGES, T), T, 0, stream>>>(dst, deg);
    k_compute_dis<<<cdiv(NPAD, T), T, 0, stream>>>(deg, dis);

    // weight / input staging (bf16, B pre-packed to fragment order)
    k_convert_x<<<NPAD, KP, 0, stream>>>(x, Abf);
    k_pack_w<<<cdiv(NTILES * 1 * 512, T), T, 0, stream>>>(W0, W0pk, IN_CH, HID, 1);
    for (int i = 0; i < 3; ++i)
        k_pack_w<<<cdiv(NTILES * 6 * 512, T), T, 0, stream>>>(
            W_h + (size_t)i * HID * HID, Whpk + (size_t)i * NTILES * 6 * 512, HID, HID, 6);
    k_pack_w<<<cdiv(NTILES * 24 * 512, T), T, 0, stream>>>(W_jk, Wjkpk, 4 * HID, HID, 24);

    dim3 ggrid(NPAD / (16 * MT), NTILES);
    // 4 GCN layers
    for (int i = 0; i < 4; ++i) {
        if (i == 0)
            k_gemm_wmma<1, KP><<<ggrid, 32, 0, stream>>>(Abf, W0pk, Hpre);
        else
            k_gemm_wmma<6, KP><<<ggrid, 32, 0, stream>>>(
                Abf, Whpk + (size_t)(i - 1) * NTILES * 6 * 512, Hpre);
        k_self_init<<<NPAD, KP, 0, stream>>>(Hpre, dis, Hagg);
        k_edge_scatter<<<cdiv(N_EDGES * 32, T), T, 0, stream>>>(src, dst, dis, Hpre, Hagg);
        const float* bias = (i == 0) ? b0 : (b_h + (size_t)(i - 1) * HID);
        k_bn_relu_emit<<<NPAD, KP, 0, stream>>>(
            Hagg, bias, gam + (size_t)i * HID, bet + (size_t)i * HID,
            mean + (size_t)i * HID, var + (size_t)i * HID, Abf, Xcat, i * HID);
    }

    // JumpingKnowledge linear: Hjk = Xcat @ W_jk  (bias folded into pooling)
    k_zero_xcat_pad<<<cdiv(NPAD * (JKW - 4 * HID), T), T, 0, stream>>>(Xcat);
    k_gemm_wmma<24, JKW><<<ggrid, 32, 0, stream>>>(Xcat, Wjkpk, Hpre);

    // global max pool + output head
    k_init_pool<<<cdiv(N_GRAPHS * HID, T), T, 0, stream>>>(keys);
    k_pool_max<<<cdiv(N_NODES * 32, T), T, 0, stream>>>(batch, Hpre, b_jk, keys);
    k_final<<<cdiv(N_GRAPHS, T), T, 0, stream>>>(keys, W_out, b_out, out);
}
